// LocalWindowAttention_4604204941621
// MI455X (gfx1250) — matrix operand order repaired and checked by reading, not compiled, not run on hardware
//
#include <hip/hip_runtime.h>

// ---------------------------------------------------------------------------
// Local window attention for gfx1250 (MI455X), bf16 WMMA with f32 accumulate.
// B=4, T=2048, C=1024, H=16, D=64, WINDOW=128
// Inputs/weights converted to bf16 once (weights pre-transposed), so all GEMM
// tiles are staged with global_load_async_to_lds_b128 (ASYNCcnt DMA) into
// double-buffered LDS while WMMAs consume the previous tile.
// ---------------------------------------------------------------------------

#define BB   4
#define TT   2048
#define CC   1024
#define HH   16
#define DD   64
#define WIN  128

typedef __attribute__((ext_vector_type(16))) __bf16 v16bf;
typedef __attribute__((ext_vector_type(4)))  __bf16 v4bf;
typedef __attribute__((ext_vector_type(8)))  float  v8f;
typedef __attribute__((ext_vector_type(4)))  int    v4i;

#if __has_builtin(__builtin_amdgcn_global_load_async_to_lds_b128) && \
    __has_builtin(__builtin_amdgcn_s_wait_asynccnt)
#define HAVE_ASYNC_LDS 1
#else
#define HAVE_ASYNC_LDS 0
#endif

#if HAVE_ASYNC_LDS
// void (v4i addrspace(1)*, v4i addrspace(3)*, imm offset, imm cpol)
__device__ __forceinline__ void async_cp16(const void* g, void* l) {
  __builtin_amdgcn_global_load_async_to_lds_b128(
      (__attribute__((address_space(1))) v4i*)((void*)g),
      (__attribute__((address_space(3))) v4i*)l, 0, 0);
}
__device__ __forceinline__ void async_wait0() {
  __builtin_amdgcn_s_wait_asynccnt(0);
}
#endif

__device__ __forceinline__ __bf16 f2bf(float x) { return (__bf16)x; }

__device__ __forceinline__ v8f vzero8() {
  v8f v;
#pragma unroll
  for (int i = 0; i < 8; ++i) v[i] = 0.0f;
  return v;
}

// A-matrix fragment (16x32 bf16), source row-major [row][ld], k contiguous.
__device__ __forceinline__ v16bf load_A_frag(const __bf16* base, int row0, int ld,
                                             int k0, int lane) {
  const int m     = lane & 15;
  const int khalf = (lane & 16) ? 16 : 0;
  const __bf16* p = base + (long)(row0 + m) * ld + k0;
  v16bf a;
#pragma unroll
  for (int e = 0; e < 16; ++e) {
    a[e]     = p[khalf + e];
  }
  return a;
}

// B-matrix fragment (32x16 bf16). Source stored TRANSPOSED: [n][ld], k contiguous.
__device__ __forceinline__ v16bf load_B_frag(const __bf16* base, int n0, int ld,
                                             int k0, int lane) {
  const int n    = lane & 15;
  const int koff = (lane & 16) ? 16 : 0;
  const __bf16* p = base + (long)(n0 + n) * ld + k0 + koff;
  v16bf b;
#pragma unroll
  for (int e = 0; e < 16; ++e) b[e] = p[e];
  return b;
}

__device__ __forceinline__ v8f wmma_bf16(v16bf a, v16bf b, v8f c) {
  return __builtin_amdgcn_wmma_f32_16x16x32_bf16(false, a, false, b,
                                                 (short)0, c, false, false);
}

// ---- async tile staging (bf16 source, contiguous 16B chunks) ---------------
__device__ __forceinline__ void stage_tileA(__bf16 (*As)[32],
                                            const __bf16* __restrict__ src,
                                            int m0, int k0, int tid) {
#pragma unroll
  for (int g = tid; g < 512; g += 128) {      // 128 rows x 32 k
    const int r = g >> 2, c8 = (g & 3) << 3;
#if HAVE_ASYNC_LDS
    async_cp16(src + (long)(m0 + r) * CC + k0 + c8, &As[r][c8]);
#else
    *(uint4*)&As[r][c8] = *(const uint4*)(src + (long)(m0 + r) * CC + k0 + c8);
#endif
  }
}

__device__ __forceinline__ void stage_tileB(__bf16 (*Bs)[32],
                                            const __bf16* __restrict__ Wt,
                                            int n0, int k0, int tid) {
#pragma unroll
  for (int g = tid; g < 256; g += 128) {      // 128 rows x 16 k
    const int r = g >> 1, c8 = (g & 1) << 3;
#if HAVE_ASYNC_LDS
    async_cp16(Wt + (long)(n0 + r) * CC + k0 + c8, &Bs[r][c8]);
#else
    *(uint4*)&Bs[r][c8] = *(const uint4*)(Wt + (long)(n0 + r) * CC + k0 + c8);
#endif
  }
}

// ---------------------------------------------------------------------------
// Kernel 0a: x fp32 -> bf16 (one pass).
// ---------------------------------------------------------------------------
__global__ __launch_bounds__(256)
void lwa_cvt_x(const float* __restrict__ src, __bf16* __restrict__ dst) {
  const long i = ((long)blockIdx.x * 256 + threadIdx.x) * 4;
  const float4 f = *(const float4*)(src + i);
  v4bf o;
  o[0] = f2bf(f.x); o[1] = f2bf(f.y); o[2] = f2bf(f.z); o[3] = f2bf(f.w);
  *(v4bf*)(dst + i) = o;
}

// ---------------------------------------------------------------------------
// Kernel 0b: weights fp32 [k][n] -> bf16 TRANSPOSED [n][k] (one pass each).
// blockIdx.y selects which of the four weights.
// ---------------------------------------------------------------------------
__global__ __launch_bounds__(256)
void lwa_cvt_w_t(const float* __restrict__ W0, const float* __restrict__ W1,
                 const float* __restrict__ W2, const float* __restrict__ W3,
                 __bf16* __restrict__ T0, __bf16* __restrict__ T1,
                 __bf16* __restrict__ T2, __bf16* __restrict__ T3) {
  const int which = blockIdx.y;
  const float* W = (which == 0) ? W0 : (which == 1) ? W1 : (which == 2) ? W2 : W3;
  __bf16*      T = (which == 0) ? T0 : (which == 1) ? T1 : (which == 2) ? T2 : T3;
  const int idx = blockIdx.x * 256 + threadIdx.x;   // 0 .. 1024*256-1
  const int n   = idx & 1023;                       // coalesced reads along n
  const int k4  = (idx >> 10) << 2;
  v4bf o;
#pragma unroll
  for (int i = 0; i < 4; ++i) o[i] = f2bf(W[(long)(k4 + i) * CC + n]);
  *(v4bf*)(T + (long)n * CC + k4) = o;
}

// ---------------------------------------------------------------------------
// Kernel 1: fused QKV projection.  Xbf[8192,1024] x Wt[1024,1024] (bf16, B
// pre-transposed) -> bf16 Q/K/V in [B,H,T,D].  blockIdx.z selects projection.
// Double-buffered async-LDS pipeline; block tile 128x64; wave tile 32x64.
// ---------------------------------------------------------------------------
__global__ __launch_bounds__(128)
void lwa_qkv_gemm(const __bf16* __restrict__ Xb,
                  const __bf16* __restrict__ Wtq, const float* __restrict__ bq,
                  const __bf16* __restrict__ Wtk, const float* __restrict__ bk,
                  const __bf16* __restrict__ Wtv, const float* __restrict__ bv,
                  __bf16* __restrict__ Qb, __bf16* __restrict__ Kb,
                  __bf16* __restrict__ Vb) {
  const int which = blockIdx.z;
  const __bf16* Wt  = (which == 0) ? Wtq : (which == 1) ? Wtk : Wtv;
  const float* bias = (which == 0) ? bq : (which == 1) ? bk : bv;
  __bf16* out       = (which == 0) ? Qb : (which == 1) ? Kb : Vb;

  const int m0   = blockIdx.y * 128;
  const int n0   = blockIdx.x * 64;
  const int tid  = threadIdx.x;
  const int lane = tid & 31;
  const int wave = tid >> 5;

  __shared__ __bf16 As[2][128][32];
  __shared__ __bf16 Bs[2][64][32];

  v8f acc[2][4];
#pragma unroll
  for (int a = 0; a < 2; ++a)
#pragma unroll
    for (int j = 0; j < 4; ++j) acc[a][j] = vzero8();

  stage_tileA(As[0], Xb, m0, 0, tid);
  stage_tileB(Bs[0], Wt, n0, 0, tid);
#if HAVE_ASYNC_LDS
  async_wait0();
#endif
  __syncthreads();

  for (int k0 = 0; k0 < CC; k0 += 32) {
    const int cur = (k0 >> 5) & 1;
    const int nxt = cur ^ 1;
    if (k0 + 32 < CC) {           // DMA next tile while computing current
      stage_tileA(As[nxt], Xb, m0, k0 + 32, tid);
      stage_tileB(Bs[nxt], Wt, n0, k0 + 32, tid);
    }

    const v16bf a0 = load_A_frag(&As[cur][0][0], wave * 32, 32, 0, lane);
    const v16bf a1 = load_A_frag(&As[cur][0][0], wave * 32 + 16, 32, 0, lane);
#pragma unroll
    for (int j = 0; j < 4; ++j) {
      const v16bf bm = load_B_frag(&Bs[cur][0][0], j * 16, 32, 0, lane);
      acc[0][j] = wmma_bf16(a0, bm, acc[0][j]);
      acc[1][j] = wmma_bf16(a1, bm, acc[1][j]);
    }
#if HAVE_ASYNC_LDS
    async_wait0();
#endif
    __syncthreads();
  }

  // Epilogue: +bias, scatter into [B,H,T,D] bf16.
  const int nlane = lane & 15;
  const int mhalf = (lane & 16) ? 8 : 0;
#pragma unroll
  for (int j = 0; j < 4; ++j) {
    const int n  = n0 + j * 16 + nlane;
    const float bn = bias[n];
    const int h = n >> 6;
    const int d = n & 63;
#pragma unroll
    for (int a = 0; a < 2; ++a) {
#pragma unroll
      for (int r = 0; r < 8; ++r) {
        const int m = m0 + wave * 32 + a * 16 + mhalf + r;
        const int b = m >> 11;
        const int t = m & 2047;
        out[(((long)(b * HH + h) * TT) + t) * DD + d] = f2bf(acc[a][j][r] + bn);
      }
    }
  }
}

// ---------------------------------------------------------------------------
// Kernel 2: banded attention.  One block per (b*H+h, 64-query tile).
// 4 waves; wave w handles 16 queries t0 = q0 + 16w attending keys
// s in [t0-128, t0+15] -> 9 key tiles inside a shared 192-key window.
// ---------------------------------------------------------------------------
__global__ __launch_bounds__(128)
void lwa_attn(const __bf16* __restrict__ Qb, const __bf16* __restrict__ Kb,
              const __bf16* __restrict__ Vb, __bf16* __restrict__ AO) {
  const int tid  = threadIdx.x;
  const int lane = tid & 31;
  const int wave = tid >> 5;
  const int bh   = blockIdx.y;       // b*H + h
  const int b    = bh >> 4;
  const int h    = bh & 15;
  const int q0   = blockIdx.x * 64;

  __shared__ __bf16 Ks[192][64];     // [s_local][c]  (same layout as global)
  __shared__ __bf16 Vs[64][224];     // [c][s_local], zero-padded to 224
  __shared__ float  Sc[4][16][160];  // per-wave scores / exp scratch
  __shared__ __bf16 Pb[4][16][160];  // per-wave probabilities (bf16, padded)

  const int s_start = q0 - WIN;      // may be negative
  const long kbase  = ((long)bh * TT + s_start) * DD;

  // ---- stage K window: 192x64 bf16 = 24 KB, contiguous in global ----------
  if (s_start >= 0) {
#if HAVE_ASYNC_LDS
#pragma unroll
    for (int g = tid; g < 1536; g += 128)                // 16B chunks
      async_cp16(Kb + kbase + (long)g * 8, (char*)&Ks[0][0] + (long)g * 16);
#else
#pragma unroll
    for (int g = tid; g < 1536; g += 128)
      *(uint4*)((char*)&Ks[0][0] + (long)g * 16) =
          *(const uint4*)(Kb + kbase + (long)g * 8);
#endif
  } else {
#pragma unroll
    for (int g = tid; g < 1536; g += 128) {
      const int sl = g >> 3;                             // 8 chunks per row
      const uint4 z = {0u, 0u, 0u, 0u};
      *(uint4*)((char*)&Ks[0][0] + (long)g * 16) =
          (s_start + sl >= 0) ? *(const uint4*)(Kb + kbase + (long)g * 8) : z;
    }
  }

  // ---- stage V transposed: Vs[c][s], padded/zeroed to s_local < 224 -------
#pragma unroll 4
  for (int g = tid; g < 64 * 56; g += 128) {
    const int c = g & 63, sb = (g >> 6) << 2;
    v4bf o;
#pragma unroll
    for (int i = 0; i < 4; ++i) {
      const int sl = sb + i, sg = s_start + sl;
      __bf16 vv = (__bf16)0.0f;
      if (sl < 192 && sg >= 0 && sg < TT) vv = Vb[((long)bh * TT + sg) * DD + c];
      o[i] = vv;
    }
    *(v4bf*)&Vs[c][sb] = o;
  }

#if HAVE_ASYNC_LDS
  if (s_start >= 0) async_wait0();
#endif
  __syncthreads();

  // ---- Q fragments straight from global (bf16, row stride D) --------------
  const int t0 = q0 + wave * 16;
  const v16bf qa0 = load_A_frag(Qb + ((long)bh * TT + t0) * DD, 0, DD, 0,  lane);
  const v16bf qa1 = load_A_frag(Qb + ((long)bh * TT + t0) * DD, 0, DD, 32, lane);

  const int nlane = lane & 15;
  const int mhalf = (lane & 16) ? 8 : 0;

  // ---- scores: 9 key tiles, K-dim 64 -> 2 WMMAs per tile -------------------
#pragma unroll
  for (int j = 0; j < 9; ++j) {
    v8f acc = vzero8();
    const v16bf bk0 = load_B_frag(&Ks[0][0], (wave + j) * 16, 64, 0,  lane);
    acc = wmma_bf16(qa0, bk0, acc);
    const v16bf bk1 = load_B_frag(&Ks[0][0], (wave + j) * 16, 64, 32, lane);
    acc = wmma_bf16(qa1, bk1, acc);
#pragma unroll
    for (int r = 0; r < 8; ++r)
      Sc[wave][mhalf + r][j * 16 + nlane] = acc[r] * 0.125f;  // 1/sqrt(64)
  }
  __syncthreads();

  // ---- masked softmax: threads 0..63 each own one query row ---------------
  if (tid < 64) {
    const int w = tid >> 4, m = tid & 15;
    float mx = -1e30f;
    for (int c = 0; c < 144; ++c) {
      const int sg = s_start + w * 16 + c;
      if (c >= m && c <= m + WIN && sg >= 0) mx = fmaxf(mx, Sc[w][m][c]);
    }
    float sum = 0.0f;
    for (int c = 0; c < 144; ++c) {
      const int sg = s_start + w * 16 + c;
      const bool ok = (c >= m) && (c <= m + WIN) && (sg >= 0);
      const float e = ok ? __expf(Sc[w][m][c] - mx) : 0.0f;
      Sc[w][m][c] = e;
      sum += e;
    }
    const float inv = 1.0f / sum;
    for (int c = 0; c < 144; ++c) Pb[w][m][c] = f2bf(Sc[w][m][c] * inv);
    for (int c = 144; c < 160; ++c) Pb[w][m][c] = (__bf16)0.0f;
  }
  __syncthreads();

  // ---- O = P @ V : K-dim 160 (padded) -> 5 steps; N = 64 -> 4 tiles --------
  v8f oacc[4];
#pragma unroll
  for (int j = 0; j < 4; ++j) oacc[j] = vzero8();

#pragma unroll
  for (int kb = 0; kb < 160; kb += 32) {
    const v16bf ap = load_A_frag(&Pb[wave][0][0], 0, 160, kb, lane);
#pragma unroll
    for (int jc = 0; jc < 4; ++jc) {
      const v16bf bv = load_B_frag(&Vs[0][0], jc * 16, 224, wave * 16 + kb, lane);
      oacc[jc] = wmma_bf16(ap, bv, oacc[jc]);
    }
  }

  // ---- write bf16 context into [B,T,C] ------------------------------------
#pragma unroll
  for (int jc = 0; jc < 4; ++jc) {
#pragma unroll
    for (int r = 0; r < 8; ++r) {
      const int t  = t0 + mhalf + r;
      const int cg = h * 64 + jc * 16 + nlane;
      AO[((long)(b * TT + t)) * CC + cg] = f2bf(oacc[jc][r]);
    }
  }
}

// ---------------------------------------------------------------------------
// Kernel 3: output projection.  AO[8192,1024] (bf16) x Wto (bf16, transposed)
// + bo -> fp32 out.  Same double-buffered async-LDS pipeline.
// ---------------------------------------------------------------------------
__global__ __launch_bounds__(128)
void lwa_out_proj(const __bf16* __restrict__ AO, const __bf16* __restrict__ Wto,
                  const float* __restrict__ bo, float* __restrict__ out) {
  const int m0   = blockIdx.y * 128;
  const int n0   = blockIdx.x * 64;
  const int tid  = threadIdx.x;
  const int lane = tid & 31;
  const int wave = tid >> 5;

  __shared__ __bf16 As[2][128][32];
  __shared__ __bf16 Bs[2][64][32];

  v8f acc[2][4];
#pragma unroll
  for (int a = 0; a < 2; ++a)
#pragma unroll
    for (int j = 0; j < 4; ++j) acc[a][j] = vzero8();

  stage_tileA(As[0], AO, m0, 0, tid);
  stage_tileB(Bs[0], Wto, n0, 0, tid);
#if HAVE_ASYNC_LDS
  async_wait0();
#endif
  __syncthreads();

  for (int k0 = 0; k0 < CC; k0 += 32) {
    const int cur = (k0 >> 5) & 1;
    const int nxt = cur ^ 1;
    if (k0 + 32 < CC) {
      stage_tileA(As[nxt], AO, m0, k0 + 32, tid);
      stage_tileB(Bs[nxt], Wto, n0, k0 + 32, tid);
    }

    const v16bf a0 = load_A_frag(&As[cur][0][0], wave * 32, 32, 0, lane);
    const v16bf a1 = load_A_frag(&As[cur][0][0], wave * 32 + 16, 32, 0, lane);
#pragma unroll
    for (int j = 0; j < 4; ++j) {
      const v16bf bm = load_B_frag(&Bs[cur][0][0], j * 16, 32, 0, lane);
      acc[0][j] = wmma_bf16(a0, bm, acc[0][j]);
      acc[1][j] = wmma_bf16(a1, bm, acc[1][j]);
    }
#if HAVE_ASYNC_LDS
    async_wait0();
#endif
    __syncthreads();
  }

  const int nlane = lane & 15;
  const int mhalf = (lane & 16) ? 8 : 0;
#pragma unroll
  for (int j = 0; j < 4; ++j) {
    const int n  = n0 + j * 16 + nlane;
    const float bn = bo[n];
#pragma unroll
    for (int a = 0; a < 2; ++a) {
#pragma unroll
      for (int r = 0; r < 8; ++r) {
        const int m = m0 + wave * 32 + a * 16 + mhalf + r;
        out[(long)m * CC + n] = acc[a][j][r] + bn;
      }
    }
  }
}

// ---------------------------------------------------------------------------
// Launch
// ---------------------------------------------------------------------------
extern "C" void kernel_launch(void* const* d_in, const int* in_sizes, int n_in,
                              void* d_out, int out_size, void* d_ws, size_t ws_size,
                              hipStream_t stream) {
  const float* x  = (const float*)d_in[0];
  const float* Wq = (const float*)d_in[1];
  const float* bq = (const float*)d_in[2];
  const float* Wk = (const float*)d_in[3];
  const float* bk = (const float*)d_in[4];
  const float* Wv = (const float*)d_in[5];
  const float* bv = (const float*)d_in[6];
  const float* Wo = (const float*)d_in[7];
  const float* bo = (const float*)d_in[8];
  float* out = (float*)d_out;

  // workspace (bf16): Q,K,V [B,H,T,D] + AO [B,T,C] + Xbf [B*T,C] + 4 Wt [C,C]
  const long NELEM = (long)BB * TT * CC;  // 8,388,608
  const long WELEM = (long)CC * CC;       // 1,048,576
  __bf16* Qb  = (__bf16*)d_ws;
  __bf16* Kb  = Qb + NELEM;
  __bf16* Vb  = Kb + NELEM;
  __bf16* AO  = Vb + NELEM;
  __bf16* Xbf = AO + NELEM;
  __bf16* Wtq = Xbf + NELEM;
  __bf16* Wtk = Wtq + WELEM;
  __bf16* Wtv = Wtk + WELEM;
  __bf16* Wto = Wtv + WELEM;              // total ~88 MB of ws

  // 0) one-time conversions: x -> bf16; weights -> bf16 transposed [n][k]
  lwa_cvt_x<<<dim3((unsigned)(NELEM / 4 / 256)), 256, 0, stream>>>(x, Xbf);
  lwa_cvt_w_t<<<dim3(1024, 4), 256, 0, stream>>>(Wq, Wk, Wv, Wo,
                                                 Wtq, Wtk, Wtv, Wto);

  // 1) QKV projections
  dim3 gq(CC / 64, (BB * TT) / 128, 3);
  lwa_qkv_gemm<<<gq, 128, 0, stream>>>(Xbf, Wtq, bq, Wtk, bk, Wtv, bv,
                                       Qb, Kb, Vb);

  // 2) banded attention
  dim3 ga(TT / 64, BB * HH, 1);
  lwa_attn<<<ga, 128, 0, stream>>>(Qb, Kb, Vb, AO);

  // 3) output projection
  dim3 go(CC / 64, (BB * TT) / 128, 1);
  lwa_out_proj<<<go, 128, 0, stream>>>(AO, Wto, bo, out);
}
